// RadianceRenderer_89713276879095
// MI455X (gfx1250) — compile-verified, matrix operand order checked by reading, not run on hardware
//
#include <hip/hip_runtime.h>
#include <math.h>

// CDNA5 / gfx1250: wave32 WMMA f16 path.
typedef __attribute__((ext_vector_type(16))) _Float16 v16h;
typedef __attribute__((ext_vector_type(8)))  float    v8f;

#define N_SAMPLES 512
#define TILES (N_SAMPLES / 16)
#define WAVES_PER_BLOCK 8
#define BOOSTER 1.0f

__device__ __forceinline__ float sigmoidf_(float x) {
    return 1.0f / (1.0f + __expf(-x));
}
__device__ __forceinline__ float softplusf_(float x) {
    return (x > 20.0f) ? x : log1pf(__expf(x));
}

// Wave-local LDS ordering: per-wave LDS region, and CDNA5 LDS ops from one
// wave complete in order (DScnt, in-order). We only need to keep the
// compiler from reordering DS ops: wavefront-scope fence + scheduling
// barrier. Emits no s_barrier — waves stay independent.
__device__ __forceinline__ void wave_lds_sync() {
    __builtin_amdgcn_fence(__ATOMIC_ACQ_REL, "wavefront");
    __builtin_amdgcn_wave_barrier();
    __builtin_amdgcn_fence(__ATOMIC_ACQ_REL, "wavefront");
}

// 16-bit operand element -> K index mapping (both A rows and B columns):
//   lane<16 : pairs j=0..3 -> K=2j,2j+1 ; j=4..7 -> K=16+2(j-4),...
//   lane>=16: same +8
__device__ __forceinline__ int kOf(int e, int half) {
    return ((((e >> 1) & 3) << 1) + ((e >> 3) << 4) + (half << 3)) + (e & 1);
}

__global__ __launch_bounds__(WAVES_PER_BLOCK * 32)
void nerf_render_wmma(const float* __restrict__ o,
                      const float* __restrict__ d,
                      const float* __restrict__ aabb,
                      const float* __restrict__ u,
                      const float* __restrict__ W1,   // [3][32]
                      const float* __restrict__ b1,   // [32]
                      const float* __restrict__ Wc,   // [32][3]
                      const float* __restrict__ bc,   // [3]
                      const float* __restrict__ Wd,   // [32][1]
                      const float* __restrict__ bd,   // [1]
                      float* __restrict__ out,        // [n_rays][4]
                      int n_rays)
{
    __shared__ float lds_h[WAVES_PER_BLOCK][16][33]; // h transpose staging (padded)
    __shared__ float lds_o[WAVES_PER_BLOCK][16][4];  // per-sample {R,G,B,sigma}

    const int lane = threadIdx.x & 31;
    const int wave = threadIdx.x >> 5;
    const int s16  = lane & 15;   // sample-in-tile / N index / A row
    const int half = lane >> 4;   // K-half selector for 16-bit operands
    const int ray  = blockIdx.x * WAVES_PER_BLOCK + wave;
    if (ray >= n_rays) return;    // uniform per wave; no block barriers used

    // ---- per-ray setup: AABB slab test (all lanes redundant, cheap) ----
    const float ox = o[ray * 3 + 0], oy = o[ray * 3 + 1], oz = o[ray * 3 + 2];
    const float dx = d[ray * 3 + 0], dy = d[ray * 3 + 1], dz = d[ray * 3 + 2];
    const float lo0 = aabb[0], lo1 = aabb[1], lo2 = aabb[2];
    const float hi0 = aabb[3], hi1 = aabb[4], hi2 = aabb[5];

    const float t1x = (lo0 - ox) / dx, t2x = (hi0 - ox) / dx;
    const float t1y = (lo1 - oy) / dy, t2y = (hi1 - oy) / dy;
    const float t1z = (lo2 - oz) / dz, t2z = (hi2 - oz) / dz;
    const float tnear = fmaxf(fmaxf(fminf(t1x, t2x),
                              fmaxf(fminf(t1y, t2y), fminf(t1z, t2z))), 0.0f);
    const float tfar  = fminf(fmaxf(t1x, t2x),
                        fminf(fmaxf(t1y, t2y), fmaxf(t1z, t2z)));
    const float act   = (tnear < tfar) ? 1.0f : 0.0f;
    const float dnorm = sqrtf(dx * dx + dy * dy + dz * dz);
    const float dt    = tfar - tnear;

    const float sx = 2.0f / (hi0 - lo0);
    const float sy = 2.0f / (hi1 - lo1);
    const float sz = 2.0f / (hi2 - lo2);

    // ---- ray-invariant WMMA B operands + biases (built once) ----
    const int n2 = s16;                       // layer-2 output column for this lane
    const int nW1lo = s16;                    // hidden unit for B1 low half
    const int nW1hi = s16 + 16;               // hidden unit for B1 high half
    const int ncl = (n2 < 2) ? n2 : 2;        // clamped Wc column (load-safe)

    v16h b1lo, b1hi, b2m;
#pragma unroll
    for (int e = 0; e < 16; ++e) {
        const int k  = kOf(e, half);          // B row (input feature / hidden idx)
        const int kc = (k < 2) ? k : 2;       // clamped W1 row (load-safe)
        const float w1l = (k < 3) ? W1[kc * 32 + nW1lo] : 0.0f;
        const float w1h = (k < 3) ? W1[kc * 32 + nW1hi] : 0.0f;
        b1lo[e] = (_Float16)w1l;
        b1hi[e] = (_Float16)w1h;
        float w2v = 0.0f;
        const float wc_v = Wc[k * 3 + ncl];
        const float wd_v = Wd[k];
        if (n2 < 3)      w2v = wc_v;
        else if (n2 == 3) w2v = wd_v;
        b2m[e] = (_Float16)w2v;
    }
    const float bias1lo = b1[nW1lo];
    const float bias1hi = b1[nW1hi];
    const float bc_c   = bc[ncl];
    const float bias2  = (n2 < 3) ? bc_c : ((n2 == 3) ? bd[0] : 0.0f);

    // ---- march the ray: 32 tiles of 16 samples ----
    float carry = 0.0f;                 // cumulative sum of sigma*delta so far
    float accR = 0.0f, accG = 0.0f, accB = 0.0f, accA = 0.0f;

    for (int tile = 0; tile < TILES; ++tile) {
        const int   s  = tile * 16 + s16;
        const float us = u[ray * N_SAMPLES + s];
        const float ts = tnear + dt * ((float)s + us) * (1.0f / N_SAMPLES);
        const int   sn = s + 1;
        const float un = u[ray * N_SAMPLES + ((sn < N_SAMPLES) ? sn : (N_SAMPLES - 1))];
        const float tn = (sn < N_SAMPLES)
                           ? (tnear + dt * ((float)sn + un) * (1.0f / N_SAMPLES))
                           : (tfar + BOOSTER);
        const float delta = (tn - ts) * dnorm;

        const float px = ox + dx * ts, py = oy + dy * ts, pz = oz + dz * ts;
        const float nx = sx * (px - lo0) - 1.0f;
        const float ny = sy * (py - lo1) - 1.0f;
        const float nz = sz * (pz - lo2) - 1.0f;
        const bool inbox = (fabsf(nx) <= 1.0f) && (fabsf(ny) <= 1.0f) && (fabsf(nz) <= 1.0f);

        // ---- layer 1: A = ndc (16x32 f16, K=0..2 live) ----
        v16h a1;
#pragma unroll
        for (int e = 0; e < 16; ++e) {
            const int k = kOf(e, half);
            const float v = (k == 0) ? nx : ((k == 1) ? ny : ((k == 2) ? nz : 0.0f));
            a1[e] = (_Float16)v;
        }
        v8f zc = {};
        v8f hlo = __builtin_amdgcn_wmma_f32_16x16x32_f16(false, a1, false, b1lo,
                                                         (short)0, zc, false, false);
        v8f hhi = __builtin_amdgcn_wmma_f32_16x16x32_f16(false, a1, false, b1hi,
                                                         (short)0, zc, false, false);

        // bias + ReLU, transpose through per-wave LDS (D: lane=col -> A2: lane=row)
#pragma unroll
        for (int v = 0; v < 8; ++v) {
            const int m = v + (half << 3);
            lds_h[wave][m][s16]      = fmaxf(hlo[v] + bias1lo, 0.0f);
            lds_h[wave][m][s16 + 16] = fmaxf(hhi[v] + bias1hi, 0.0f);
        }
        wave_lds_sync();

        v16h a2;
#pragma unroll
        for (int e = 0; e < 16; ++e) {
            a2[e] = (_Float16)lds_h[wave][s16][kOf(e, half)];
        }
        v8f d2 = __builtin_amdgcn_wmma_f32_16x16x32_f16(false, a2, false, b2m,
                                                        (short)0, zc, false, false);

        // activations; scatter {R,G,B,sigma} per sample into LDS
        if (n2 < 4) {
#pragma unroll
            for (int v = 0; v < 8; ++v) {
                const int m = v + (half << 3);
                const float val = d2[v] + bias2;
                lds_o[wave][m][n2] = (n2 < 3) ? sigmoidf_(val) : softplusf_(val);
            }
        }
        wave_lds_sync();

        // ---- compositing: 16-lane prefix scan of sigma*delta + carry ----
        const float cr = lds_o[wave][s16][0];
        const float cg = lds_o[wave][s16][1];
        const float cb = lds_o[wave][s16][2];
        const float sg = lds_o[wave][s16][3];
        const float sd = inbox ? (sg * delta) : 0.0f;

        float incl = sd;
#pragma unroll
        for (int off = 1; off < 16; off <<= 1) {
            const float t = __shfl_up(incl, off, 16);
            if (s16 >= off) incl += t;
        }
        const float excl = incl - sd;
        const float Tcur = __expf(-(carry + excl));   // transmittance before sample
        const float w    = Tcur * (1.0f - __expf(-sd));
        accR += w * cr;  accG += w * cg;  accB += w * cb;  accA += w;
        carry += __shfl(incl, 15, 16);                // tile total, broadcast

        wave_lds_sync();   // reads complete before next tile's LDS writes (in-order anyway)
    }

    // ---- reduce the 16 per-lane partials, write [R,G,B,alpha] * active ----
#pragma unroll
    for (int off = 8; off >= 1; off >>= 1) {
        accR += __shfl_xor(accR, off, 16);
        accG += __shfl_xor(accG, off, 16);
        accB += __shfl_xor(accB, off, 16);
        accA += __shfl_xor(accA, off, 16);
    }
    if (lane == 0) {
        float4 res;
        res.x = accR * act;
        res.y = accG * act;
        res.z = accB * act;
        res.w = accA * act;
        *reinterpret_cast<float4*>(out + (size_t)ray * 4) = res;
    }
}

extern "C" void kernel_launch(void* const* d_in, const int* in_sizes, int n_in,
                              void* d_out, int out_size, void* d_ws, size_t ws_size,
                              hipStream_t stream) {
    (void)n_in; (void)out_size; (void)d_ws; (void)ws_size;
    const float* o    = (const float*)d_in[0];
    const float* d    = (const float*)d_in[1];
    const float* aabb = (const float*)d_in[2];
    const float* u    = (const float*)d_in[3];
    const float* W1   = (const float*)d_in[4];
    const float* b1   = (const float*)d_in[5];
    const float* Wc   = (const float*)d_in[6];
    const float* bc   = (const float*)d_in[7];
    const float* Wd   = (const float*)d_in[8];
    const float* bd   = (const float*)d_in[9];
    float* out = (float*)d_out;

    const int n_rays = in_sizes[0] / 3;
    dim3 block(WAVES_PER_BLOCK * 32);
    dim3 grid((n_rays + WAVES_PER_BLOCK - 1) / WAVES_PER_BLOCK);
    hipLaunchKernelGGL(nerf_render_wmma, grid, block, 0, stream,
                       o, d, aabb, u, W1, b1, Wc, bc, Wd, bd, out, n_rays);
}
